// DyConv_45226005627024
// MI455X (gfx1250) — compile-verified, matrix operand order checked by reading, not run on hardware
//
#include <hip/hip_runtime.h>
#include <hip/hip_bf16.h>

// ---------------------------------------------------------------------------
// DyConv (CondConv) for MI455X / gfx1250, wave32 + WMMA f16->f32.
//   x:      [16, 256, 64, 64] f32
//   weight: [4, 256, 256, 3, 3] f32
//   bias:   [4, 256] f32
//   fc1_w:  [4, 256], fc1_b: [4], fc2_w: [4,4], fc2_b: [4]
//   out:    [16, 256, 64, 64] f32
// Conv as implicit GEMM per batch: M=256(O), N=4096(HW), K=2304(C*9).
// ---------------------------------------------------------------------------

typedef __attribute__((ext_vector_type(16))) _Float16 v16h;
typedef __attribute__((ext_vector_type(8)))  _Float16 v8h;
typedef __attribute__((ext_vector_type(4)))  _Float16 v4h;
typedef __attribute__((ext_vector_type(8)))  float    v8f;
typedef __attribute__((ext_vector_type(4)))  int      v4i_;

#define B_   16
#define C_   256
#define O_   256
#define K_   4
#define H_   64
#define W_   64
#define HW_  4096
#define KD_  2304      // C*9 (GEMM K dimension)
#define PW_  66        // padded width/height
#define PP_  4356      // 66*66 padded plane elements

// --- CDNA5 async global->LDS copy (ASYNCcnt path), with safe fallback -------
#if __has_builtin(__builtin_amdgcn_global_load_async_to_lds_b128)
#define HAVE_ASYNC_LDS 1
#endif

__device__ __forceinline__ void copy16_g2l(const _Float16* __restrict__ g,
                                           _Float16* __restrict__ l) {
#if defined(HAVE_ASYNC_LDS)
    __builtin_amdgcn_global_load_async_to_lds_b128(
        (__attribute__((address_space(1))) v4i_*)(void*)g,
        (__attribute__((address_space(3))) v4i_*)(void*)l, 0, 0);
#else
    *(uint4*)l = *(const uint4*)g;
#endif
}

__device__ __forceinline__ void wait_async_lds() {
#if defined(HAVE_ASYNC_LDS)
#if __has_builtin(__builtin_amdgcn_s_wait_asynccnt)
    __builtin_amdgcn_s_wait_asynccnt(0);
#else
    asm volatile("s_wait_asynccnt 0x0" ::: "memory");
#endif
#endif
}

// ---------------------------------------------------------------------------
// 1) Per-(b,c) plane stats: mean and rsqrt(var + eps).
// ---------------------------------------------------------------------------
__global__ __launch_bounds__(256) void dyc_stats(const float* __restrict__ x,
                                                 float* __restrict__ mean,
                                                 float* __restrict__ rsig) {
    const int plane = blockIdx.x;                 // b*C + c
    const float* p = x + (size_t)plane * HW_;
    float s = 0.f, sq = 0.f;
    for (int i = threadIdx.x; i < HW_; i += 256) {
        float v = p[i];
        s += v; sq += v * v;
    }
    __shared__ float ss[256], sqq[256];
    ss[threadIdx.x] = s; sqq[threadIdx.x] = sq;
    __syncthreads();
    for (int st = 128; st > 0; st >>= 1) {
        if (threadIdx.x < st) {
            ss[threadIdx.x]  += ss[threadIdx.x + st];
            sqq[threadIdx.x] += sqq[threadIdx.x + st];
        }
        __syncthreads();
    }
    if (threadIdx.x == 0) {
        float m = ss[0] * (1.f / HW_);
        float v = sqq[0] * (1.f / HW_) - m * m;
        mean[plane] = m;
        rsig[plane] = rsqrtf(v + 1e-5f);
    }
}

// ---------------------------------------------------------------------------
// 2) Attention MLP + softmax over K experts. One wave32 per sample.
// ---------------------------------------------------------------------------
__global__ __launch_bounds__(32) void dyc_attn(const float* __restrict__ mean,
                                               const float* __restrict__ fc1w,
                                               const float* __restrict__ fc1b,
                                               const float* __restrict__ fc2w,
                                               const float* __restrict__ fc2b,
                                               float* __restrict__ attn) {
    const int b = blockIdx.x;
    const int lane = threadIdx.x;                 // 0..31 (wave32)
    float h[K_];
    for (int k = 0; k < K_; ++k) {
        float s = 0.f;
        for (int c = lane; c < C_; c += 32)
            s += mean[b * C_ + c] * fc1w[k * C_ + c];
        for (int off = 16; off > 0; off >>= 1)
            s += __shfl_down(s, off, 32);
        s = __shfl(s, 0, 32);
        h[k] = fmaxf(s + fc1b[k], 0.f);
    }
    if (lane == 0) {
        float lg[K_], mx = -1e30f;
        for (int k = 0; k < K_; ++k) {
            float t = fc2b[k];
            for (int j = 0; j < K_; ++j) t += fc2w[k * K_ + j] * h[j];
            lg[k] = t; mx = fmaxf(mx, t);
        }
        float se = 0.f;
        for (int k = 0; k < K_; ++k) { lg[k] = __expf(lg[k] - mx); se += lg[k]; }
        float inv = 1.f / se;
        for (int k = 0; k < K_; ++k) attn[b * K_ + k] = lg[k] * inv;
    }
}

// ---------------------------------------------------------------------------
// 3) Per-sample aggregated bias: aggb[b][o] = sum_k attn[b][k] * bias[k][o]
// ---------------------------------------------------------------------------
__global__ __launch_bounds__(256) void dyc_aggb(const float* __restrict__ attn,
                                                const float* __restrict__ bias,
                                                float* __restrict__ aggb) {
    const int b = blockIdx.x, o = threadIdx.x;
    float s = 0.f;
    for (int k = 0; k < K_; ++k)
        s += attn[b * K_ + k] * bias[k * O_ + o];
    aggb[b * O_ + o] = s;
}

// ---------------------------------------------------------------------------
// 4) Normalized, zero-padded f16 activations: xnp[b][c][66][66]
// ---------------------------------------------------------------------------
__global__ __launch_bounds__(256) void dyc_xnp(const float* __restrict__ x,
                                               const float* __restrict__ mean,
                                               const float* __restrict__ rsig,
                                               _Float16* __restrict__ xnp) {
    const int plane = blockIdx.x;                 // b*C + c
    const float mu = mean[plane], rs = rsig[plane];
    const float* src = x + (size_t)plane * HW_;
    _Float16* dst = xnp + (size_t)plane * PP_;
    for (int i = threadIdx.x; i < PP_; i += 256) dst[i] = (_Float16)0.f;
    __syncthreads();
    for (int i = threadIdx.x; i < HW_; i += 256) {
        int h = i >> 6, w = i & 63;
        dst[(h + 1) * PW_ + (w + 1)] = (_Float16)((src[i] - mu) * rs);
    }
}

// ---------------------------------------------------------------------------
// 5) Per-sample mixed weights in f16: aggw[b][o][c*9+rs] (row length 2304)
//    weight layout [k][o][c][3][3] is already [o][2304]-contiguous per k.
// ---------------------------------------------------------------------------
__global__ __launch_bounds__(256) void dyc_aggw(const float* __restrict__ attn,
                                                const float* __restrict__ weight,
                                                _Float16* __restrict__ aggw) {
    const int o = blockIdx.x, b = blockIdx.y;
    float a0 = attn[b * K_ + 0], a1 = attn[b * K_ + 1];
    float a2 = attn[b * K_ + 2], a3 = attn[b * K_ + 3];
    const size_t wrow = (size_t)o * KD_;
    const size_t kstep = (size_t)O_ * KD_;
    _Float16* dst = aggw + ((size_t)b * O_ + o) * KD_;
    for (int i = threadIdx.x; i < KD_; i += 256) {
        float v = a0 * weight[wrow + i]
                + a1 * weight[kstep + wrow + i]
                + a2 * weight[2 * kstep + wrow + i]
                + a3 * weight[3 * kstep + wrow + i];
        dst[i] = (_Float16)v;
    }
}

// ---------------------------------------------------------------------------
// 6) Implicit-GEMM conv. grid = (2 Mtiles, 32 row-pairs, 16 batches),
//    block = 256 (8 waves, wave grid 4(M) x 2(N)).
//    Block tile: 128(M) x 128(N = two image rows) x 32(K).
//    Per wave: 2 A-frags x 4 B-frags -> 8 accumulators, 8 WMMAs per K-step.
//    A tile fed by global_load_async_to_lds_b128 (ASYNCcnt); B tile is the
//    im2col gather, filled as per-thread 4(k)x4(n) micro-tiles so the
//    transposed [n][k] LDS image is written with b64 stores.
// ---------------------------------------------------------------------------
__global__ __launch_bounds__(256) void dyc_gemm(const _Float16* __restrict__ aggw,
                                                const _Float16* __restrict__ xnp,
                                                const float* __restrict__ aggb,
                                                float* __restrict__ out) {
    const int mt = blockIdx.x;                    // 0..1  (M tile of 128)
    const int h0 = blockIdx.y * 2;                // first of two output rows
    const int b  = blockIdx.z;

    __shared__ _Float16 As[128 * 32];             // [m][k] row-major, 8KB
    __shared__ _Float16 Bs[128 * 32];             // [n][k] row-major (transposed), 8KB

    const int t    = threadIdx.x;
    const int lane = t & 31;
    const int wv   = t >> 5;                      // 0..7
    const int wm   = wv >> 1;                     // 0..3 -> M offset
    const int wn   = wv & 1;                      // 0..1 -> N offset
    const int m0   = wm * 32;
    const int nW   = wn * 64;
    const int lo16 = lane & 15;
    const int hi   = (lane >= 16) ? 8 : 0;        // K-half select per frag layout
    const int madd = (lane >= 16) ? 8 : 0;

    v8f acc[2][4] = {};                           // [M sub][N sub] 16x16 tiles

    const _Float16* Ab = aggw + ((size_t)b * O_ + (size_t)mt * 128) * KD_;
    const _Float16* Xb = xnp + (size_t)b * C_ * PP_;

    // A-tile slot for this thread: 512 x 16B chunks, 2 per thread.
    const int arow = t >> 1;                      // rows 0..127
    const int ac40 = (t & 1) * 2;                 // chunk 0/2
    const int ac41 = (t & 1) * 2 + 1;             // chunk 1/3

    // B-tile micro-tile for this thread: 4 K rows x 4 pixels.
    const int kg   = (t >> 5) * 4;                // local K base: 0,4,...,28
    const int ng   = (t & 31) * 4;                // pixel base: 0..124 (4-chunks
    const int brow = ng >> 6;                     //   never cross an image row)
    const int w0   = ng & 63;

    for (int k0 = 0; k0 < KD_; k0 += 32) {
        __syncthreads();
        // --- A tile via async global->LDS (ASYNCcnt).
        copy16_g2l(Ab + (size_t)arow * KD_ + k0 + ac40 * 8, &As[arow * 32 + ac40 * 8]);
        copy16_g2l(Ab + (size_t)arow * KD_ + k0 + ac41 * 8, &As[arow * 32 + ac41 * 8]);
        if (k0 + 32 < KD_)                        // prefetch next K-step's A chunk
            __builtin_prefetch(Ab + (size_t)arow * KD_ + k0 + 32, 0, 1);

        // --- B tile (im2col): row kk -> c=kk/9, r=(kk%9)/3, s=(kk%9)%3.
        //     Gather 4x4 (k x n), transpose in registers, b64-store into [n][k].
        {
            _Float16 tr[4][4];
            #pragma unroll
            for (int dk = 0; dk < 4; ++dk) {
                int kk  = k0 + kg + dk;
                int c   = kk / 9;
                int rr  = kk - c * 9;
                int r   = rr / 3;
                int sft = rr - r * 3;
                const _Float16* src =
                    Xb + (size_t)c * PP_ + (h0 + brow + r) * PW_ + sft + w0;
                #pragma unroll
                for (int dn = 0; dn < 4; ++dn) tr[dn][dk] = src[dn];
            }
            #pragma unroll
            for (int dn = 0; dn < 4; ++dn) {
                v4h vkk = {tr[dn][0], tr[dn][1], tr[dn][2], tr[dn][3]};
                *(v4h*)(&Bs[(ng + dn) * 32 + kg]) = vkk;  // 8B aligned
            }
        }
        wait_async_lds();                         // this wave's A chunks resident
        __syncthreads();                          // whole tile visible to all waves

        // --- A fragments (16x32 f16): lanes 0-15 K{0..7,16..23}, 16-31 +8.
        v16h a[2];
        #pragma unroll
        for (int aidx = 0; aidx < 2; ++aidx) {
            const int mrow = m0 + aidx * 16 + lo16;
            v8h alo = *(const v8h*)(&As[mrow * 32 + hi]);
            v8h ahi = *(const v8h*)(&As[mrow * 32 + 16 + hi]);
            a[aidx] = __builtin_shufflevector(alo, ahi,
                        0,1,2,3,4,5,6,7,8,9,10,11,12,13,14,15);
        }
        #pragma unroll
        for (int j = 0; j < 4; ++j) {
            const int nrow = nW + j * 16 + lo16;
            v8h blo = *(const v8h*)(&Bs[nrow * 32 + hi]);
            v8h bhi = *(const v8h*)(&Bs[nrow * 32 + 16 + hi]);
            v16h bf = __builtin_shufflevector(blo, bhi,
                        0,1,2,3,4,5,6,7,8,9,10,11,12,13,14,15);
            acc[0][j] = __builtin_amdgcn_wmma_f32_16x16x32_f16(
                            false, a[0], false, bf, (short)0, acc[0][j], false, false);
            acc[1][j] = __builtin_amdgcn_wmma_f32_16x16x32_f16(
                            false, a[1], false, bf, (short)0, acc[1][j], false, false);
        }
    }

    // --- Epilogue: C/D layout = VGPR v: lanes 0-15 -> M=v, lanes 16-31 -> M=v+8.
    #pragma unroll
    for (int aidx = 0; aidx < 2; ++aidx) {
        #pragma unroll
        for (int j = 0; j < 4; ++j) {
            const int n  = nW + j * 16 + lo16;
            const int hh = h0 + (n >> 6);
            const int w  = n & 63;
            #pragma unroll
            for (int v = 0; v < 8; ++v) {
                const int o = mt * 128 + m0 + aidx * 16 + v + madd;
                out[(((size_t)b * O_ + o) * H_ + hh) * W_ + w] =
                    acc[aidx][j][v] + aggb[b * O_ + o];
            }
        }
    }
}

// ---------------------------------------------------------------------------
// Launcher. Workspace carve-out (bytes):
//   mean  f32[4096]     @ 0
//   rsig  f32[4096]     @ 16384
//   attn  f32[64]       @ 32768
//   aggb  f32[4096]     @ 36864
//   aggw  f16[9437184]  @ 65536          (18,874,368 B)
//   xnp   f16[17842176] @ 18,939,904    (35,684,352 B)   total ~54.6 MB
// ---------------------------------------------------------------------------
extern "C" void kernel_launch(void* const* d_in, const int* in_sizes, int n_in,
                              void* d_out, int out_size, void* d_ws, size_t ws_size,
                              hipStream_t stream) {
    const float* x      = (const float*)d_in[0];
    const float* weight = (const float*)d_in[1];
    const float* bias   = (const float*)d_in[2];
    const float* fc1w   = (const float*)d_in[3];
    const float* fc1b   = (const float*)d_in[4];
    const float* fc2w   = (const float*)d_in[5];
    const float* fc2b   = (const float*)d_in[6];
    float* out = (float*)d_out;

    char* ws = (char*)d_ws;
    float*    mean = (float*)(ws + 0);
    float*    rsig = (float*)(ws + 16384);
    float*    attn = (float*)(ws + 32768);
    float*    aggb = (float*)(ws + 36864);
    _Float16* aggw = (_Float16*)(ws + 65536);
    _Float16* xnp  = (_Float16*)(ws + 18939904ull);

    dyc_stats<<<B_ * C_, 256, 0, stream>>>(x, mean, rsig);
    dyc_attn <<<B_, 32, 0, stream>>>(mean, fc1w, fc1b, fc2w, fc2b, attn);
    dyc_aggb <<<B_, 256, 0, stream>>>(attn, bias, aggb);
    dyc_xnp  <<<B_ * C_, 256, 0, stream>>>(x, mean, rsig, xnp);
    dyc_aggw <<<dim3(O_, B_), 256, 0, stream>>>(attn, weight, aggw);
    dyc_gemm <<<dim3(2, 32, B_), 256, 0, stream>>>(aggw, xnp, aggb, out);
}
